// Attention_85676007621236
// MI455X (gfx1250) — compile-verified, matrix operand order checked by reading, not run on hardware
//
#include <hip/hip_runtime.h>
#include <hip/hip_bf16.h>

// ---------------------------------------------------------------------------
// Types for CDNA5 WMMA (wave32): v_wmma_f32_16x16x32_bf16
// ---------------------------------------------------------------------------
typedef __attribute__((ext_vector_type(16))) __bf16 v16bf;
typedef __attribute__((ext_vector_type(8)))  float  v8f;

union FragBF16 {
    v16bf v;
    uint4 q[2];
};

// Round-to-nearest-even f32 -> bf16 (raw u16)
__device__ __forceinline__ unsigned short f2bf(float f) {
    unsigned int u = __float_as_uint(f);
    u += 0x7FFFu + ((u >> 16) & 1u);
    return (unsigned short)(u >> 16);
}

// Load one 16x32 bf16 A/B fragment for this lane.
// Per CDNA5 ISA 7.12.2 (16-bit matrix 16x32): lanes 0-15 hold K=0..7
// (VGPR0-3) and K=16..23 (VGPR4-7); lanes 16-31 hold K=8..15 and K=24..31.
// Both runs are 8 contiguous bf16 = one 16-byte (b128) load each.
__device__ __forceinline__ v16bf load_frag(const unsigned short* rowPtr, int half) {
    FragBF16 f;
    f.q[0] = *(const uint4*)(rowPtr + 8 * half);
    f.q[1] = *(const uint4*)(rowPtr + 16 + 8 * half);
    return f.v;
}

#define WMMA_BF16(A, B, C) \
    __builtin_amdgcn_wmma_f32_16x16x32_bf16(false, (A), false, (B), (short)0, (C), false, false)

// ---------------------------------------------------------------------------
// Elementwise f32 -> bf16 conversion, 4 elements / thread
// ---------------------------------------------------------------------------
__global__ void cvt_f32_to_bf16_x4(const float* __restrict__ in,
                                   unsigned short* __restrict__ out, int n4) {
    int i = blockIdx.x * blockDim.x + threadIdx.x;
    if (i >= n4) return;
    float4 f = ((const float4*)in)[i];
    unsigned int lo = (unsigned int)f2bf(f.x) | ((unsigned int)f2bf(f.y) << 16);
    unsigned int hi = (unsigned int)f2bf(f.z) | ((unsigned int)f2bf(f.w) << 16);
    ((uint2*)out)[i] = make_uint2(lo, hi);
}

// ---------------------------------------------------------------------------
// QKV projection with 2x2 register blocking: each wave computes a 32x32
// output block (4 WMMAs per 8 b128 loads per k-step).
//   out[row=b*1024+n, o] = sum_c x[row,c] * w_qkv[o,c]
//   M=16384, K=1024, Ncols=3072.
//   Q,K (o<2048)  -> qk  [B*N, 2048]   (d-contiguous per head)
//   V  (o>=2048)  -> vT  [B*H, 64, N]  (key-contiguous: transposed)
// ---------------------------------------------------------------------------
__global__ void qkv_gemm_bf16(const unsigned short* __restrict__ X,
                              const unsigned short* __restrict__ W,
                              unsigned short* __restrict__ qk,
                              unsigned short* __restrict__ vT) {
    const int lane = threadIdx.x & 31;
    const int wave = threadIdx.x >> 5;
    const int unit = blockIdx.x * 8 + wave;        // 32x32 block index
    const int BN = 3072 / 32;                      // 96
    if (unit >= 512 * BN) return;                  // 16384/32 = 512
    const int bm = unit / BN;
    const int bn = unit % BN;
    const int half = lane >> 4, l15 = lane & 15;

    const unsigned short* aRow0 = X + (size_t)(bm * 32 + l15) * 1024;
    const unsigned short* aRow1 = aRow0 + (size_t)16 * 1024;
    const unsigned short* bRow0 = W + (size_t)(bn * 32 + l15) * 1024;
    const unsigned short* bRow1 = bRow0 + (size_t)16 * 1024;

    v8f c00 = {}, c01 = {}, c10 = {}, c11 = {};
    #pragma unroll 2
    for (int k0 = 0; k0 < 1024; k0 += 32) {
        const v16bf a0 = load_frag(aRow0 + k0, half);
        const v16bf a1 = load_frag(aRow1 + k0, half);
        const v16bf b0 = load_frag(bRow0 + k0, half);
        const v16bf b1 = load_frag(bRow1 + k0, half);
        c00 = WMMA_BF16(a0, b0, c00);
        c01 = WMMA_BF16(a0, b1, c01);
        c10 = WMMA_BF16(a1, b0, c10);
        c11 = WMMA_BF16(a1, b1, c11);
    }

    // Store: D layout per sub-tile: VGPR r -> M = r + 8*half, N = l15.
    // Q/K vs V boundary (o == 2048) is 16-aligned -> uniform per sub-tile.
    v8f acc[4] = {c00, c01, c10, c11};
    #pragma unroll
    for (int mi = 0; mi < 2; ++mi) {
        #pragma unroll
        for (int ni = 0; ni < 2; ++ni) {
            const v8f c = acc[mi * 2 + ni];
            const int rowBase = bm * 32 + mi * 16 + 8 * half;
            const int oBase   = bn * 32 + ni * 16;      // wave-uniform
            if (oBase < 2048) {
                const int o = oBase + l15;
                #pragma unroll
                for (int r = 0; r < 8; ++r)
                    qk[(size_t)(rowBase + r) * 2048 + o] = f2bf(c[r]);
            } else {
                const int hd = oBase + l15 - 2048;      // h*64 + d
                #pragma unroll
                for (int r = 0; r < 8; ++r) {
                    const int row = rowBase + r;
                    const int b = row >> 10, n = row & 1023;
                    // (b*16+h)*64 + d == b*1024 + hd
                    vT[(((size_t)b * 1024 + hd) << 10) + n] = f2bf(c[r]);
                }
            }
        }
    }
}

// ---------------------------------------------------------------------------
// Fused attention for one (b, h, 16-row query block).
//   grid = B*H*(N/16) = 16384 blocks, 256 threads (8 waves).
//   Phase 1: S = QK^T * scale  -> 64 KB LDS (8 waves x 8 key tiles)
//   Phase 2: block softmax, P repacked in-place to bf16 in LDS
//   Phase 3: O = P @ V, split-K over wave pairs so all 8 waves run WMMAs
// ---------------------------------------------------------------------------
__global__ void attn_fused(const unsigned short* __restrict__ qk,
                           const unsigned short* __restrict__ vT,
                           unsigned short* __restrict__ O) {
    __shared__ float s[16 * 1024];                 // 64 KB, reused as bf16 P
    __shared__ float red[16 * 16];
    __shared__ float rowmax[16];
    __shared__ float rowsum[16];
    __shared__ float part[4 * 256];                // split-K partials (4 KB)
    unsigned short* p = (unsigned short*)s;

    const int t    = threadIdx.x;
    const int lane = t & 31, wave = t >> 5;
    const int half = lane >> 4, l15 = lane & 15;

    const int blk = blockIdx.x;
    const int rb  = blk & 63;                      // query row-block
    const int bh  = blk >> 6;                      // b*16 + h
    const int b   = bh >> 4, h = bh & 15;

    const unsigned short* Qbase =
        qk + ((size_t)(b * 1024 + rb * 16) * 2048) + h * 64;
    const unsigned short* Kbase =
        qk + ((size_t)b * 1024) * 2048 + 1024 + h * 64;

    // Q fragments for this wave's 16 query rows (reused for all key tiles)
    const unsigned short* qRow = Qbase + (size_t)l15 * 2048;
    const v16bf qf0 = load_frag(qRow + 0,  half);
    const v16bf qf1 = load_frag(qRow + 32, half);

    // --- Phase 1: S = (Q K^T) * scale, 64 key tiles over 8 waves ----------
    for (int ct = wave; ct < 64; ct += 8) {
        const unsigned short* kRow = Kbase + (size_t)(ct * 16 + l15) * 2048;
        const v16bf kf0 = load_frag(kRow + 0,  half);
        const v16bf kf1 = load_frag(kRow + 32, half);
        v8f c = {};
        c = WMMA_BF16(qf0, kf0, c);
        c = WMMA_BF16(qf1, kf1, c);
        #pragma unroll
        for (int r = 0; r < 8; ++r)
            s[(r + 8 * half) * 1024 + ct * 16 + l15] = c[r] * 0.125f; // 1/sqrt(64)
    }
    __syncthreads();

    // --- Phase 2: softmax; thread owns row = t>>4, cols (t&15) + 16*i -----
    const int row = t >> 4, cpart = t & 15;
    float ev[64];
    float m = -3.0e38f;
    #pragma unroll
    for (int i = 0; i < 64; ++i) {
        const float v = s[row * 1024 + cpart + 16 * i];
        ev[i] = v;
        m = fmaxf(m, v);
    }
    red[row * 16 + cpart] = m;
    __syncthreads();
    if (cpart == 0) {
        float mm = red[row * 16];
        #pragma unroll
        for (int i = 1; i < 16; ++i) mm = fmaxf(mm, red[row * 16 + i]);
        rowmax[row] = mm;
    }
    __syncthreads();
    const float rm = rowmax[row];
    float sum = 0.f;
    #pragma unroll
    for (int i = 0; i < 64; ++i) { ev[i] = __expf(ev[i] - rm); sum += ev[i]; }
    red[row * 16 + cpart] = sum;
    __syncthreads();
    if (cpart == 0) {
        float ss = 0.f;
        #pragma unroll
        for (int i = 0; i < 16; ++i) ss += red[row * 16 + i];
        rowsum[row] = ss;
    }
    __syncthreads();
    const float inv = 1.0f / rowsum[row];
    // all reads of s[] completed before the barriers above; safe to repack
    #pragma unroll
    for (int i = 0; i < 64; ++i)
        p[row * 1024 + cpart + 16 * i] = f2bf(ev[i] * inv);
    __syncthreads();

    // --- Phase 3: O(16x64) = P(16x1024) @ V(1024x64) ----------------------
    // 4 d-tiles x 2 K-halves: wave w -> d-tile (w&3), K-half (w>>2).
    const int dblock = (wave & 3) * 16;
    const int khalf  = wave >> 2;
    const unsigned short* pRow = p + l15 * 1024;                  // LDS
    const unsigned short* vRow =
        vT + (((size_t)bh * 64 + dblock + l15) << 10);            // key-contiguous
    v8f c = {};
    const int kbeg = khalf * 512;
    #pragma unroll 4
    for (int k0 = kbeg; k0 < kbeg + 512; k0 += 32) {
        const v16bf a  = load_frag(pRow + k0, half);              // ds_load_b128
        const v16bf bf = load_frag(vRow + k0, half);              // global_load_b128
        c = WMMA_BF16(a, bf, c);
    }
    if (khalf == 1) {
        #pragma unroll
        for (int r = 0; r < 8; ++r)
            part[(wave & 3) * 256 + (r + 8 * half) * 16 + l15] = c[r];
    }
    __syncthreads();
    if (khalf == 0) {
        #pragma unroll
        for (int r = 0; r < 8; ++r) {
            const int mrow = r + 8 * half;
            const float v = c[r] + part[(wave & 3) * 256 + mrow * 16 + l15];
            const int n = rb * 16 + mrow;
            const int d = dblock + l15;
            O[((size_t)(b * 1024 + n) << 10) + h * 64 + d] = f2bf(v);
        }
    }
}

// ---------------------------------------------------------------------------
// Output projection with 2x2 register blocking, fp32 output + bias.
//   out[row, o] = sum_c A[row,c] * w_proj[o,c] + b_proj[o]
//   M=16384, K=1024, N=1024.
// ---------------------------------------------------------------------------
__global__ void proj_gemm_bf16(const unsigned short* __restrict__ A,
                               const unsigned short* __restrict__ W,
                               const float* __restrict__ bias,
                               float* __restrict__ out) {
    const int lane = threadIdx.x & 31;
    const int wave = threadIdx.x >> 5;
    const int unit = blockIdx.x * 8 + wave;
    const int BN = 1024 / 32;                      // 32
    if (unit >= 512 * BN) return;
    const int bm = unit / BN;
    const int bn = unit % BN;
    const int half = lane >> 4, l15 = lane & 15;

    const unsigned short* aRow0 = A + (size_t)(bm * 32 + l15) * 1024;
    const unsigned short* aRow1 = aRow0 + (size_t)16 * 1024;
    const unsigned short* bRow0 = W + (size_t)(bn * 32 + l15) * 1024;
    const unsigned short* bRow1 = bRow0 + (size_t)16 * 1024;

    v8f c00 = {}, c01 = {}, c10 = {}, c11 = {};
    #pragma unroll 2
    for (int k0 = 0; k0 < 1024; k0 += 32) {
        const v16bf a0 = load_frag(aRow0 + k0, half);
        const v16bf a1 = load_frag(aRow1 + k0, half);
        const v16bf b0 = load_frag(bRow0 + k0, half);
        const v16bf b1 = load_frag(bRow1 + k0, half);
        c00 = WMMA_BF16(a0, b0, c00);
        c01 = WMMA_BF16(a0, b1, c01);
        c10 = WMMA_BF16(a1, b0, c10);
        c11 = WMMA_BF16(a1, b1, c11);
    }

    v8f acc[4] = {c00, c01, c10, c11};
    #pragma unroll
    for (int mi = 0; mi < 2; ++mi) {
        #pragma unroll
        for (int ni = 0; ni < 2; ++ni) {
            const v8f c = acc[mi * 2 + ni];
            const int rowBase = bm * 32 + mi * 16 + 8 * half;
            const int col     = bn * 32 + ni * 16 + l15;
            const float bv = bias[col];
            #pragma unroll
            for (int r = 0; r < 8; ++r)
                out[(size_t)(rowBase + r) * 1024 + col] = c[r] + bv;
        }
    }
}

// ---------------------------------------------------------------------------
// Launch
// ---------------------------------------------------------------------------
extern "C" void kernel_launch(void* const* d_in, const int* in_sizes, int n_in,
                              void* d_out, int out_size, void* d_ws, size_t ws_size,
                              hipStream_t stream) {
    (void)in_sizes; (void)n_in; (void)out_size; (void)ws_size;

    const float* x      = (const float*)d_in[0];   // [16,1024,1024]
    const float* w_qkv  = (const float*)d_in[1];   // [3072,1024]
    const float* w_proj = (const float*)d_in[2];   // [1024,1024]
    const float* b_proj = (const float*)d_in[3];   // [1024]
    float* out = (float*)d_out;                    // [16,1024,1024]

    // Workspace layout (bf16 as raw u16), total ~168 MB
    unsigned short* xb     = (unsigned short*)d_ws;
    unsigned short* wqkvb  = xb     + (size_t)16384 * 1024;        //  32 MB
    unsigned short* wprojb = wqkvb  + (size_t)3072  * 1024;        //   6 MB
    unsigned short* qkbuf  = wprojb + (size_t)1024  * 1024;        //   2 MB
    unsigned short* vTbuf  = qkbuf  + (size_t)16384 * 2048;        //  64 MB
    unsigned short* obuf   = vTbuf  + (size_t)16    * 1024 * 1024; //  32 MB (+32 MB)

    // 1) fp32 -> bf16 conversions
    {
        int n4 = (16384 * 1024) / 4;
        cvt_f32_to_bf16_x4<<<(n4 + 255) / 256, 256, 0, stream>>>(x, xb, n4);
        n4 = (3072 * 1024) / 4;
        cvt_f32_to_bf16_x4<<<(n4 + 255) / 256, 256, 0, stream>>>(w_qkv, wqkvb, n4);
        n4 = (1024 * 1024) / 4;
        cvt_f32_to_bf16_x4<<<(n4 + 255) / 256, 256, 0, stream>>>(w_proj, wprojb, n4);
    }

    // 2) QKV projection: 512 x 96 wave-units (32x32 each), 8 waves/block
    {
        const int units = 512 * 96;
        qkv_gemm_bf16<<<units / 8, 256, 0, stream>>>(xb, wqkvb, qkbuf, vTbuf);
    }

    // 3) fused attention: B*H*(N/16) = 16384 blocks
    attn_fused<<<16 * 16 * 64, 256, 0, stream>>>(qkbuf, vTbuf, obuf);

    // 4) output projection: 512 x 32 wave-units, 8 waves/block
    {
        const int units = 512 * 32;
        proj_gemm_bf16<<<units / 8, 256, 0, stream>>>(obuf, wprojb, b_proj, out);
    }
}
